// Attention_GATE_6176162972115
// MI455X (gfx1250) — matrix operand order repaired and checked by reading, not compiled, not run on hardware
//
#include <hip/hip_runtime.h>
#include <hip/hip_bf16.h>
#include <stdint.h>

#define Bsz 8
#define Cc  512
#define Ee  512
#define Wd  32
#define Pp  16384
#define NEGV (-1e30f)

typedef __attribute__((ext_vector_type(16))) __bf16        v16bf;
typedef __attribute__((ext_vector_type(8)))  float         v8f;
typedef __attribute__((ext_vector_type(8)))  unsigned int  v8u;
typedef __attribute__((ext_vector_type(4)))  unsigned int  u32x4;

static __device__ __forceinline__ __bf16 f2bf(float f) { return (__bf16)f; }

static __device__ __forceinline__ v8f wmma_bf16(v16bf a, v16bf b, v8f c) {
    return __builtin_amdgcn_wmma_f32_16x16x32_bf16(false, a, false, b, (short)0, c, false, false);
}

// Build a v16bf fragment from two 16B vector loads (dwords lo[0..3], hi[0..3]).
static __device__ __forceinline__ v16bf frag_from_2xb128(u32x4 lo, u32x4 hi) {
    v8u u;
    u[0] = lo[0]; u[1] = lo[1]; u[2] = lo[2]; u[3] = lo[3];
    u[4] = hi[0]; u[5] = hi[1]; u[6] = hi[2]; u[7] = hi[3];
    return __builtin_bit_cast(v16bf, u);
}

// -------------------------------------------------------------------------
// Kernel 1: projections.
//   objectT stored as objTw[b][w][c]  (bf16, A-operand layout for phase 1)
//   colorT  stored as colTc[b][c][w]  (bf16, w contiguous -> b128 B-frags)
//   value = sum_e Wm[c][e] * emb[b][e][w]
// grid = B*2*32 blocks of 64 threads (2 waves; wave = w-half n-tile)
// -------------------------------------------------------------------------
__global__ void __launch_bounds__(64) proj_kernel(
    const float* __restrict__ W_obj, const float* __restrict__ W_col,
    const float* __restrict__ obj_emb, const float* __restrict__ col_emb,
    __bf16* __restrict__ objTw, __bf16* __restrict__ colTc)
{
    const int lane  = threadIdx.x & 31;
    const int ntile = threadIdx.x >> 5;            // 0..1 (w halves)
    const int ctile = blockIdx.x & 31;             // c tile
    const int mat   = (blockIdx.x >> 5) & 1;       // 0=object 1=color
    const int bb    = blockIdx.x >> 6;
    const float* Wm  = mat ? W_col   : W_obj;
    const float* emb = mat ? col_emb : obj_emb;

    const int h   = lane >> 4;
    const int l16 = lane & 15;
    const int mrow = ctile * 16 + l16;             // A row (c)
    const int n    = ntile * 16 + l16;             // B col (w)

    v8f acc = {};
    for (int kc = 0; kc < Ee; kc += 32) {
        v16bf a, b;
#pragma unroll
        for (int v = 0; v < 8; ++v) {
            const int k0 = kc + 16 * h + 2 * v;
            a[2 * v]     = f2bf(Wm[mrow * Ee + k0]);
            a[2 * v + 1] = f2bf(Wm[mrow * Ee + k0 + 1]);
        }
#pragma unroll
        for (int v = 0; v < 8; ++v) {
            const int k = kc + 16 * h + 2 * v;
            b[2 * v]     = f2bf(emb[(bb * Ee + k) * Wd + n]);
            b[2 * v + 1] = f2bf(emb[(bb * Ee + k + 1) * Wd + n]);
        }
        acc = wmma_bf16(a, b, acc);
    }
    // D slot v: (m = ctile*16 + v + 8h, n = w)
#pragma unroll
    for (int v = 0; v < 8; ++v) {
        const int m = ctile * 16 + v + 8 * h;
        if (mat) colTc[(bb * Cc + m) * Wd + n] = f2bf(acc[v]);   // [b][c][w]
        else     objTw[(bb * Wd + n) * Cc + m] = f2bf(acc[v]);   // [b][w][c]
    }
}

// -------------------------------------------------------------------------
// Kernel 2: fused attn pipeline per 16-pixel tile (1 wave per tile)
//   phase1: attnT(w,p)  = objTw(w,c) x target(c,p)       [32 bf16 WMMAs]
//   phase2: attn2T(v,p) = o2cT(v,w)  x attnT(w,p)        [2 WMMAs, via LDS]
//   mask + word-softmax, stash exp(attn_word) -> eAttn[b][w][p]
//   phase3: outWC(p,c)  = attn_word(p,w) x colTc(w,c)    [32 WMMAs]
// grid = 1024 blocks x 256 threads (8 waves)
// -------------------------------------------------------------------------
__global__ void __launch_bounds__(256) attn_kernel(
    const float* __restrict__ target,        // B,C,P
    const float* __restrict__ o2c,           // B,W,W
    const unsigned char* __restrict__ mask,  // B,W (bool)
    const __bf16* __restrict__ objTw,        // B,W,C
    const __bf16* __restrict__ colTc,        // B,C,W
    float* __restrict__ outWC,               // B,P,C
    float* __restrict__ eAttn)               // B,W,P
{
    __shared__ float sm[8 * 16 * 36];        // per-wave 16x32 tile, stride 36
    const int lane = threadIdx.x & 31;
    const int wv   = threadIdx.x >> 5;
    const int tile = blockIdx.x * 8 + wv;    // 0..8191
    const int bb   = tile >> 10;
    const int p0   = (tile & 1023) << 4;
    const int h    = lane >> 4;
    const int l16  = lane & 15;
    float* smw = sm + wv * (16 * 36);

    // ---- phase 1: attnT = objTw x target, K = C = 512 ----
    v8f acc0 = {}, acc1 = {};
    for (int kc = 0; kc < Cc; kc += 32) {
        v16bf bfr;
#pragma unroll
        for (int v = 0; v < 8; ++v) {
            const int k = kc + ((v < 4) ? 2 * v : 16 + 2 * (v - 4)) + 8 * h;
            const float* src = target + (size_t)(bb * Cc + k) * Pp + p0 + l16;
            bfr[2 * v]     = f2bf(src[0]);
            bfr[2 * v + 1] = f2bf(src[Pp]);
        }
        // A-fragments: two contiguous 16B runs per fragment -> 2x b128 each
        const u32x4* a0p = (const u32x4*)(objTw + (size_t)(bb * Wd + l16) * Cc + kc);
        const u32x4* a1p = (const u32x4*)((const __bf16*)a0p + 16 * Cc);
        // dword runs: [4h .. 4h+3] and [8+4h .. 8+4h+3]  (u32x4 index h and 2+h)
        v16bf a0 = frag_from_2xb128(a0p[h], a0p[2 + h]);
        v16bf a1 = frag_from_2xb128(a1p[h], a1p[2 + h]);
        acc0 = wmma_bf16(a0, bfr, acc0);
        acc1 = wmma_bf16(a1, bfr, acc1);
    }
    // spill attnT to LDS as [p][w]
#pragma unroll
    for (int v = 0; v < 8; ++v) {
        smw[l16 * 36 + (v + 8 * h)]      = acc0[v];
        smw[l16 * 36 + (16 + v + 8 * h)] = acc1[v];
    }

    // ---- phase 2: attn2T = o2cT x attnT, K = W = 32 ----
    v16bf aT0, aT1, bL;
#pragma unroll
    for (int v = 0; v < 8; ++v) {
        const int k0 = ((v < 4) ? 2 * v : 16 + 2 * (v - 4)) + 8 * h;
        const float* ob = o2c + bb * (Wd * Wd);
        aT0[2 * v]     = f2bf(ob[k0 * Wd + l16]);
        aT0[2 * v + 1] = f2bf(ob[(k0 + 1) * Wd + l16]);
        aT1[2 * v]     = f2bf(ob[k0 * Wd + 16 + l16]);
        aT1[2 * v + 1] = f2bf(ob[(k0 + 1) * Wd + 16 + l16]);
    }
#pragma unroll
    for (int v = 0; v < 8; ++v) {        // contiguous 8B-aligned pairs -> ds_load_b64
        const int k = ((v < 4) ? 2 * v : 16 + 2 * (v - 4)) + 8 * h;
        const float2 pr = *(const float2*)(smw + l16 * 36 + k);
        bL[2 * v]     = f2bf(pr.x);
        bL[2 * v + 1] = f2bf(pr.y);
    }
    v8f z = {};
    v8f at0 = wmma_bf16(aT0, bL, z);     // words 0..15  (m = v+8h)
    v8f at1 = wmma_bf16(aT1, bL, z);     // words 16..31

    // ---- mask + softmax over 32 words for pixel p = p0+l16 ----
    const unsigned char mv = mask[bb * Wd + lane];
    const unsigned maskBits = (unsigned)__ballot(mv != 0);
    float x0[8], x1[8];
#pragma unroll
    for (int v = 0; v < 8; ++v) {
        const int w0 = v + 8 * h, w1 = 16 + v + 8 * h;
        x0[v] = ((maskBits >> w0) & 1u) ? NEGV : at0[v];
        x1[v] = ((maskBits >> w1) & 1u) ? NEGV : at1[v];
    }
    float mx = x0[0];
#pragma unroll
    for (int v = 0; v < 8; ++v) { mx = fmaxf(mx, x0[v]); mx = fmaxf(mx, x1[v]); }
    mx = fmaxf(mx, __shfl_xor(mx, 16));
    float e0[8], e1[8], s = 0.f;
#pragma unroll
    for (int v = 0; v < 8; ++v) {
        e0[v] = __expf(x0[v] - mx); e1[v] = __expf(x1[v] - mx);
        s += e0[v] + e1[v];
    }
    s += __shfl_xor(s, 16);
    const float inv = __frcp_rn(s);
    float aw0[8], aw1[8];
#pragma unroll
    for (int v = 0; v < 8; ++v) { aw0[v] = e0[v] * inv; aw1[v] = e1[v] * inv; }

    // stash exp(attn_word) for the pixel-softmax (layout [b][w][p], coalesced)
#pragma unroll
    for (int v = 0; v < 8; ++v) {
        eAttn[(size_t)(bb * Wd + (v + 8 * h)) * Pp + p0 + l16]      = __expf(aw0[v]);
        eAttn[(size_t)(bb * Wd + (16 + v + 8 * h)) * Pp + p0 + l16] = __expf(aw1[v]);
    }

    // ---- phase 3 A-frag: attn_word as (p x w) bf16 — pure in-lane repack ----
    v16bf aA;
#pragma unroll
    for (int av = 0; av < 4; ++av) {
        aA[2 * av]     = f2bf(aw0[2 * av]);
        aA[2 * av + 1] = f2bf(aw0[2 * av + 1]);
    }
#pragma unroll
    for (int av = 4; av < 8; ++av) {
        aA[2 * av]     = f2bf(aw1[2 * (av - 4)]);
        aA[2 * av + 1] = f2bf(aw1[2 * (av - 4) + 1]);
    }

    // ---- phase 3: outWC tile = attn_word x colorT, 32 N-tiles of c ----
    for (int nt = 0; nt < 32; ++nt) {
        const int n0 = nt * 16;
        // B-frag: lane (h,n) needs bf16 elements [8h .. 8h+7] and [16+8h .. 16+8h+7] of row c=n
        //         -> two 16B runs = two b128 loads
        const u32x4* cp = (const u32x4*)(colTc + (size_t)(bb * Cc + n0 + l16) * Wd);
        v16bf bC = frag_from_2xb128(cp[h], cp[2 + h]);
        v8f zc = {};
        v8f dc = wmma_bf16(aA, bC, zc);
#pragma unroll
        for (int v = 0; v < 8; ++v) {
            const int m = v + 8 * h;    // pixel within tile
            outWC[(size_t)(bb * Pp + p0 + m) * Cc + n0 + l16] = dc[v];
        }
    }
}

// -------------------------------------------------------------------------
// Kernel 3: sumExp[b*W+w] = sum_p eAttn[b][w][p]   (deterministic reduction)
// -------------------------------------------------------------------------
__global__ void __launch_bounds__(256) sumexp_kernel(
    const float* __restrict__ eAttn, float* __restrict__ sumExp)
{
    __shared__ float red[256];
    const int row = blockIdx.x;                 // b*W + w
    const float* src = eAttn + (size_t)row * Pp;
    float s = 0.f;
    for (int i = threadIdx.x; i < Pp; i += 256) s += src[i];
    red[threadIdx.x] = s;
    __syncthreads();
    for (int o = 128; o > 0; o >>= 1) {
        if (threadIdx.x < o) red[threadIdx.x] += red[threadIdx.x + o];
        __syncthreads();
    }
    if (threadIdx.x == 0) sumExp[row] = red[0];
}

// -------------------------------------------------------------------------
// Kernel 4: gate[b][p] = sigmoid(784 * sum_w (W_mask[w]/S[b,w])*e[b,w,p]
//                                - sum_w W_mask[w] + b_mask)
// -------------------------------------------------------------------------
__global__ void __launch_bounds__(256) gate_kernel(
    const float* __restrict__ eAttn, const float* __restrict__ sumExp,
    const float* __restrict__ W_mask, const float* __restrict__ b_mask,
    float* __restrict__ gate)
{
    __shared__ float coeff[Wd];
    __shared__ float sW;
    const int bb = blockIdx.x >> 6;             // 64 blocks per batch
    const int p  = ((blockIdx.x & 63) << 8) + threadIdx.x;
    if (threadIdx.x < Wd)
        coeff[threadIdx.x] = W_mask[threadIdx.x] / sumExp[bb * Wd + threadIdx.x];
    if (threadIdx.x == 0) {
        float t = 0.f;
        for (int w = 0; w < Wd; ++w) t += W_mask[w];
        sW = t;
    }
    __syncthreads();
    float acc = 0.f;
#pragma unroll 8
    for (int w = 0; w < Wd; ++w)
        acc += coeff[w] * eAttn[(size_t)(bb * Wd + w) * Pp + p];
    const float x = 784.0f * acc - sW + b_mask[0];
    gate[bb * Pp + p] = 1.0f / (1.0f + __expf(-x));
}

// -------------------------------------------------------------------------
extern "C" void kernel_launch(void* const* d_in, const int* in_sizes, int n_in,
                              void* d_out, int out_size, void* d_ws, size_t ws_size,
                              hipStream_t stream)
{
    const float* input      = (const float*)d_in[0];   // B,C,IH,IW
    const float* color_emb  = (const float*)d_in[1];   // B,E,W
    const float* object_emb = (const float*)d_in[2];   // B,E,W
    const float* obj2color  = (const float*)d_in[3];   // B,W,W
    const unsigned char* mask = (const unsigned char*)d_in[4]; // B,W bool
    const float* W_obj      = (const float*)d_in[5];   // C,E
    const float* W_col      = (const float*)d_in[6];   // C,E
    const float* W_mask     = (const float*)d_in[7];   // 1,W
    const float* b_mask     = (const float*)d_in[8];   // 1

    char* ws = (char*)d_ws;
    __bf16* objTw  = (__bf16*)ws;                              // 256 KB
    __bf16* colTc  = (__bf16*)(ws + 262144);                   // 256 KB
    float*  eAttn  = (float*)(ws + 524288);                    // 16 MB
    float*  sumExp = (float*)(ws + 524288 + (size_t)Bsz * Wd * Pp * 4);

    float* outWC = (float*)d_out;                              // B,P,C
    float* gate  = outWC + (size_t)Bsz * Pp * Cc;              // B,1,IH,IW

    proj_kernel<<<Bsz * 2 * 32, 64, 0, stream>>>(W_obj, W_col, object_emb,
                                                 color_emb, objTw, colTc);
    attn_kernel<<<(Bsz * Pp / 16) / 8, 256, 0, stream>>>(input, obj2color, mask,
                                                         objTw, colTc, outWC, eAttn);
    sumexp_kernel<<<Bsz * Wd, 256, 0, stream>>>(eAttn, sumExp);
    gate_kernel<<<Bsz * (Pp / 256), 256, 0, stream>>>(eAttn, sumExp, W_mask,
                                                      b_mask, gate);
}
